// GeneralizedRingAttractorNoGain_ref_23321672417826
// MI455X (gfx1250) — matrix operand order repaired and checked by reading, not compiled, not run on hardware
//
#include <hip/hip_runtime.h>

#define Bsz   128
#define Ssz   1024
#define Ksig  4
#define Nn    256
#define BT    16                 // batch rows per workgroup (= WMMA M)
#define NTILES (Nn/16)           // 16 column tiles of 16
#define KBLKS  (Nn/32)           // 8 k-blocks of 32 (bf16 WMMA K)
#define P1TILES (KBLKS*5*NTILES) // 640 phase-1 packed tiles
#define KBHALVES (5*NTILES*512)  // 40960 halves = 80KB per k-block stage
#define P2BASE  (P1TILES*512)    // phase-2 (W_delta7) packed base offset

typedef __attribute__((ext_vector_type(16))) __bf16 v16bf;
typedef __attribute__((ext_vector_type(8)))  __bf16 v8bf;
typedef __attribute__((ext_vector_type(8)))  float  v8f;
typedef __attribute__((ext_vector_type(4)))  unsigned int v4u;
typedef __attribute__((ext_vector_type(8)))  int v8i;
typedef __attribute__((ext_vector_type(4)))  int v4i;

__device__ __forceinline__ unsigned short f2bf(float f) {
    unsigned u = __float_as_uint(f);
    u += 0x7FFFu + ((u >> 16) & 1u);            // round-to-nearest-even
    return (unsigned short)(u >> 16);
}

// One TDM descriptor: contiguous 80KB (10240 x 8B) global -> LDS.
__device__ __forceinline__ void tdm_load_kb(const unsigned short* gsrc, unsigned lds_addr) {
    unsigned long long ga = (unsigned long long)(uintptr_t)gsrc;
    v4u g0 = { 1u,                                           // count=1, user load
               lds_addr,                                     // D#.lds_addr
               (unsigned)(ga & 0xffffffffu),                 // global_addr[31:0]
               (unsigned)((ga >> 32) & 0x01ffffffu) | (2u << 30) }; // addr[56:32] | type=2
    v8i g1 = { (int)(3u << 16),        // workgroup_mask=0 | data_size=3 (8 bytes)
               (int)(0x2800u << 16),   // tensor_dim0 = 10240 (low16 @ [63:48])
               (int)(1u << 16),        // tensor_dim0 hi16=0 | tensor_dim1 = 1
               (int)(0x2800u << 16),   // tensor_dim1 hi16=0 | tile_dim0 = 10240
               0,                      // tile_dim1 = 0 (1-D), tile_dim2 = 0
               10240,                  // tensor_dim0_stride low32
               0, 0 };
    v4i g2 = {0, 0, 0, 0};
    v4i g3 = {0, 0, 0, 0};
    v8i g4 = {0, 0, 0, 0, 0, 0, 0, 0};
    __builtin_amdgcn_tensor_load_to_lds(g0, g1, g2, g3, g4, 0);
}

// Prepack into CDNA5 bf16 B-operand layout. Phase-1 mats ordered [kb][mat][nt][512]
// so each k-block's whole-WG working set is one contiguous 80KB TDM transfer.
// Phase-2 (W_delta7) ordered [nt][kb][512] for direct global loads.
__global__ void prep_weights(const float* __restrict__ Wo,
                             const float* __restrict__ Wa,
                             unsigned short* __restrict__ packed) {
    int p      = blockIdx.x * blockDim.x + threadIdx.x;
    int within = p & 511;
    int tile   = p >> 9;
    int lane   = within >> 4;
    int q      = within & 15;
    int kb, nt, mat;
    if (tile < P1TILES) {
        kb = tile / 80; int rem = tile % 80; mat = rem >> 4; nt = rem & 15;
    } else {
        int t2 = tile - P1TILES; nt = t2 >> 3; kb = t2 & 7; mat = 5;
    }
    int k = kb * 32 + ((lane >= 16) ? 8 : 0) + (q & 7) + ((q >= 8) ? 16 : 0);
    int n = nt * 16 + (lane & 15);
    float w;
    if (mat == 0)      w = -0.1f + 0.1f * Wo[n * Nn + k];            // J0 + J1*Wo
    else if (mat <= 4) w = Wa[((mat - 1) * Nn + n) * Nn + k];
    else               w = __cosf(6.283185307179586f * (float)(n - k) * (1.0f / (float)Nn));
    packed[p] = f2bf(w);
}

__global__ void __launch_bounds__(256, 1)
ring_attractor(const float* __restrict__ act,          // (B,S,K)
               const float* __restrict__ r_init,       // (B,N)
               const unsigned short* __restrict__ packed,
               float* __restrict__ out_rd7,            // (B,S,N)
               float* __restrict__ out_bump) {         // (B,S,N)
    __shared__ __align__(32) unsigned short stage[2][KBHALVES];   // 2 x 80KB TDM staging
    __shared__ __align__(32) float          r_f[BT * Nn];         // f32 recurrent state
    __shared__ __align__(32) unsigned short rb[2][BT * Nn];       // bf16 ping-pong copies
    __shared__ __align__(32) float          rd7_s[BT * Nn];
    __shared__ float At_s[BT * Ksig];
    __shared__ float red_s[BT * 16];
    __shared__ float inv_s[BT];

    const int tid  = threadIdx.x;
    const int lane = tid & 31;
    const int wave = tid >> 5;                 // 0..7
    const int b0   = blockIdx.x * BT;
    const unsigned stage_lds0 = (unsigned)(uintptr_t)(void*)&stage[0][0];
    const unsigned stage_lds1 = (unsigned)(uintptr_t)(void*)&stage[1][0];

    for (int i = tid; i < BT * Nn; i += 256) {
        float v  = r_init[(long)(b0 + (i >> 8)) * Nn + (i & 255)];
        r_f[i]   = v;
        rb[0][i] = f2bf(v);
    }
    __syncthreads();

    // 16-bit A layout: lanes 0-15 hold K 0-7 & 16-23, lanes 16-31 hold K 8-15 & 24-31
    const int m_a  = lane & 15;
    const int sel  = (lane >> 4) << 3;
    // C/D layout: lane<16 -> n=lane, m=v ; lane>=16 -> n=lane-16, m=v+8
    const int mrow = (lane >> 4) << 3;
    const int ncol = lane & 15;

    for (int t = 0; t < Ssz; ++t) {
        const int cur = t & 1, nxt = cur ^ 1;

        if (tid < BT * Ksig) {
            int m = tid >> 2, a = tid & 3;
            At_s[tid] = act[((long)(b0 + m) * Ssz + t) * Ksig + a];
        }
        __syncthreads();                               // step boundary

        // prime the TDM pipeline for this step
        if (wave == 0) tdm_load_kb(packed, stage_lds0);

        // ---- phase 1: rec = r@M^T + sum_a At[:,a]*(r@Wa[a]^T) ----
        v8f acc[2][5] = {};
        for (int kb = 0; kb < KBLKS; ++kb) {
            if (wave == 0) __builtin_amdgcn_s_wait_tensorcnt(0);   // TDM(kb) complete
            __syncthreads();                                       // release stage[kb&1]
            if (wave == 0 && kb + 1 < KBLKS)                       // overlap DMA with compute
                tdm_load_kb(packed + (long)(kb + 1) * KBHALVES,
                            (kb & 1) ? stage_lds0 : stage_lds1);

            v8bf alo = *(const v8bf*)&rb[cur][m_a * Nn + kb * 32 + sel];
            v8bf ahi = *(const v8bf*)&rb[cur][m_a * Nn + kb * 32 + 16 + sel];
            v16bf a;
#pragma unroll
            for (int i = 0; i < 8; ++i) { a[i] = alo[i]; a[8 + i] = ahi[i]; }

            const unsigned short* sb = &stage[kb & 1][0];
#pragma unroll
            for (int wtl = 0; wtl < 2; ++wtl) {
                const int nt = wave * 2 + wtl;
                const unsigned short* sp = sb + nt * 512 + lane * 16;
#pragma unroll
                for (int mat = 0; mat < 5; ++mat) {
                    v16bf b = *(const v16bf*)(sp + mat * (NTILES * 512));
                    acc[wtl][mat] = __builtin_amdgcn_wmma_f32_16x16x32_bf16(
                        false, a, false, b, (short)0, acc[wtl][mat], false, false);
                }
            }
        }

        // combine + tanh + alpha-blend
#pragma unroll
        for (int wtl = 0; wtl < 2; ++wtl) {
            const int nt = wave * 2 + wtl;
#pragma unroll
            for (int v = 0; v < 8; ++v) {
                const int m = v + mrow;
                const int n = nt * 16 + ncol;
                const float* Am = &At_s[m * 4];
                float rec = acc[wtl][0][v] + Am[0] * acc[wtl][1][v] + Am[1] * acc[wtl][2][v]
                                           + Am[2] * acc[wtl][3][v] + Am[3] * acc[wtl][4][v];
                rec = tanhf(rec);
                float rnew = r_f[m * Nn + n] * 0.85f + 0.15f * rec;
                r_f[m * Nn + n]     = rnew;
                rb[nxt][m * Nn + n] = f2bf(rnew);
            }
        }
        __syncthreads();

        // ---- phase 2: rd7 = r_new @ W_delta7 (L2-resident direct loads) ----
#pragma unroll
        for (int wtl = 0; wtl < 2; ++wtl) {
            const int nt = wave * 2 + wtl;
            v8f accd = {};
            for (int kb = 0; kb < KBLKS; ++kb) {
                v8bf alo = *(const v8bf*)&rb[nxt][m_a * Nn + kb * 32 + sel];
                v8bf ahi = *(const v8bf*)&rb[nxt][m_a * Nn + kb * 32 + 16 + sel];
                v16bf a;
#pragma unroll
                for (int i = 0; i < 8; ++i) { a[i] = alo[i]; a[8 + i] = ahi[i]; }
                const unsigned short* wp =
                    packed + P2BASE + ((nt * KBLKS + kb) * 512) + lane * 16;
                v16bf bd = *(const v16bf*)(wp);
                accd = __builtin_amdgcn_wmma_f32_16x16x32_bf16(
                    false, a, false, bd, (short)0, accd, false, false);
            }
#pragma unroll
            for (int v = 0; v < 8; ++v)
                rd7_s[(v + mrow) * Nn + nt * 16 + ncol] = accd[v];
        }
        __syncthreads();

        // ---- row max, normalize, coalesced stores of both outputs ----
        {
            const int m = tid >> 4;
            const int c = (tid & 15) * 16;
            float pmax = rd7_s[m * Nn + c];
#pragma unroll
            for (int j = 1; j < 16; ++j) pmax = fmaxf(pmax, rd7_s[m * Nn + c + j]);
            red_s[tid] = pmax;
            __syncthreads();
            if (tid < BT) {
                float mx = red_s[tid * 16];
#pragma unroll
                for (int j = 1; j < 16; ++j) mx = fmaxf(mx, red_s[tid * 16 + j]);
                inv_s[tid] = 1.0f / mx;
            }
            __syncthreads();
            const float inv  = inv_s[m];
            const long  base = ((long)(b0 + m) * Ssz + t) * Nn + c;
#pragma unroll
            for (int j = 0; j < 16; ++j) {
                out_rd7[base + j]  = rd7_s[m * Nn + c + j] * inv;
                out_bump[base + j] = r_f[m * Nn + c + j];
            }
        }
    }
}

extern "C" void kernel_launch(void* const* d_in, const int* in_sizes, int n_in,
                              void* d_out, int out_size, void* d_ws, size_t ws_size,
                              hipStream_t stream) {
    const float* act = (const float*)d_in[0];   // (B,S,K)
    const float* r0  = (const float*)d_in[1];   // (B,N)
    const float* Wo  = (const float*)d_in[2];   // (N,N)
    const float* Wa  = (const float*)d_in[3];   // (K,N,N)
    unsigned short* packed = (unsigned short*)d_ws;       // 768 KB packed bf16 weights
    float* out1 = (float*)d_out;                          // rd7_hist
    float* out2 = out1 + (long)Bsz * Ssz * Nn;            // bump_hist

    const int total = (P1TILES + NTILES * KBLKS) * 512;   // 393216 packed halves
    prep_weights<<<total / 256, 256, 0, stream>>>(Wo, Wa, packed);
    ring_attractor<<<Bsz / BT, 256, 0, stream>>>(act, r0, packed, out1, out2);
    (void)in_sizes; (void)n_in; (void)out_size; (void)ws_size;
}
